// ImportancePooling_3908420239562
// MI455X (gfx1250) — compile-verified, matrix operand order checked by reading, not run on hardware
//
#include <hip/hip_runtime.h>
#include <hip/hip_bf16.h>

typedef __attribute__((ext_vector_type(16))) _Float16 v16h;
typedef __attribute__((ext_vector_type(4)))  _Float16 v4h;
typedef __attribute__((ext_vector_type(8)))  float    v8f;
typedef __attribute__((ext_vector_type(4)))  unsigned int v4u;   // 16B, align 16

#define DIM   256
#define HID   128
#define KN    32
#define A_STR 264   // f16 row stride (132 dwords) -> staggers LDS banks for A-frag loads
#define H_STR 132   // f32 row stride for h

// Build a v16h WMMA operand from two 16-byte dword-vector loads.
union FragCast {
    v4u  u[2];
    v16h h;
};

// ---------------------------------------------------------------------------
// Prepass: W1 [256][128] f32 (row-major) -> W1t [128][256] f16 (row-major,
// i.e. transposed) so each WMMA B-fragment lane reads two contiguous 16B runs.
// ---------------------------------------------------------------------------
__global__ void w1_to_f16_t_kernel(const float* __restrict__ W1,
                                   _Float16* __restrict__ W1t) {
    int i = blockIdx.x * blockDim.x + threadIdx.x;   // 0 .. 32767
    int n = i >> 8;      // hidden unit 0..127
    int k = i & 255;     // input dim   0..255
    W1t[i] = (_Float16)W1[k * HID + n];
}

// ---------------------------------------------------------------------------
// One block per batch element b. 256 threads = 8 wave32.
// ---------------------------------------------------------------------------
__global__ __launch_bounds__(256)
void importance_pool_kernel(const float*      __restrict__ x,
                            const long long*  __restrict__ nidx,
                            const float*      __restrict__ pw,
                            const _Float16*   __restrict__ W1t,
                            const float*      __restrict__ b1,
                            const float*      __restrict__ W2,
                            const float*      __restrict__ b2,
                            float*            __restrict__ out) {
    __shared__ __align__(16) float nf32[KN][DIM];                     // 32 KB
    __shared__ __align__(16) union { _Float16 a[KN][A_STR]; float h[KN][H_STR]; } sh; // 16.5 KB
    __shared__ int   sidx[KN];
    __shared__ float wmix[KN];

    const int b    = blockIdx.x;
    const int tid  = threadIdx.x;
    const int lane = tid & 31;
    const int wave = tid >> 5;

    if (tid < KN) sidx[tid] = (int)nidx[(long long)b * KN + tid];
    __syncthreads();

    // ---- Gather 32 rows x 256 f32 (x fits in 192MB L2 -> mostly L2 hits).
    // 2048 float4 chunks over 256 threads: 8 each. Also drop an f16 copy.
    for (int i = 0; i < 8; ++i) {
        int c   = tid + 256 * i;
        int row = c >> 6;            // 0..31
        int col = (c & 63) << 2;     // 0..252, multiple of 4
        const float4 v = *(const float4*)(x + (long long)sidx[row] * DIM + col);
        *(float4*)&nf32[row][col] = v;
        v4h hv = { (_Float16)v.x, (_Float16)v.y, (_Float16)v.z, (_Float16)v.w };
        *(v4h*)&sh.a[row][col] = hv;                 // packed 8B ds_store
    }
    __syncthreads();

    // ---- WMMA GEMM: nf[32x256](f16) x W1t^T[256x128](f16) -> h[32x128](f32)
    // wave = N-tile (8 tiles of 16 cols); each wave does both M-tiles.
    const int nt   = wave;
    const int m    = lane & 15;
    const int half = lane >> 4;           // 0: K 0-7,16-23; 1: K 8-15,24-31 of each 32-step
    v8f acc0 = {}; v8f acc1 = {};
    const _Float16* wrow = W1t + (nt * 16 + m) * DIM;

    #pragma unroll
    for (int k0 = 0; k0 < DIM; k0 += 32) {
        const int kb = k0 + half * 8;    // kb*2 bytes: multiple of 16
        // A fragments (rows m and 16+m), per documented 16-bit A layout:
        // halves 0-7 = K[kb..kb+7], halves 8-15 = K[kb+16..kb+23]
        FragCast A0, A1, B;
        A0.u[0] = *(const v4u*)&sh.a[m][kb];
        A0.u[1] = *(const v4u*)&sh.a[m][kb + 16];
        A1.u[0] = *(const v4u*)&sh.a[16 + m][kb];
        A1.u[1] = *(const v4u*)&sh.a[16 + m][kb + 16];
        // B fragment: column nt*16+m of W1, same K striping (mirror layout)
        B.u[0]  = *(const v4u*)(wrow + kb);
        B.u[1]  = *(const v4u*)(wrow + kb + 16);
        acc0 = __builtin_amdgcn_wmma_f32_16x16x32_f16(false, A0.h, false, B.h, (short)0, acc0, false, false);
        acc1 = __builtin_amdgcn_wmma_f32_16x16x32_f16(false, A1.h, false, B.h, (short)0, acc1, false, false);
    }
    __syncthreads();   // everyone done reading sh.a before we overwrite with h

    // ---- bias + ReLU, scatter C fragments (lane = col nt*16+m; vgpr r = row)
    const float bias = b1[nt * 16 + m];
    #pragma unroll
    for (int r = 0; r < 8; ++r) {
        const int row = r + half * 8;
        sh.h[row][nt * 16 + m]      = fmaxf(acc0[r] + bias, 0.0f);
        sh.h[16 + row][nt * 16 + m] = fmaxf(acc1[r] + bias, 0.0f);
    }
    __syncthreads();

    // ---- logits + dual softmax + mix (wave 0 only; EXEC full within wave)
    if (tid < KN) {
        float lg = b2[0];
        #pragma unroll 4
        for (int n = 0; n < HID; ++n) lg = fmaf(sh.h[tid][n], W2[n], lg);

        float mx = lg;
        #pragma unroll
        for (int off = 16; off >= 1; off >>= 1) mx = fmaxf(mx, __shfl_xor(mx, off, 32));
        float e = __expf(lg - mx);
        float s = e;
        #pragma unroll
        for (int off = 16; off >= 1; off >>= 1) s += __shfl_xor(s, off, 32);
        const float learned = e / s;

        const float pv = pw[(long long)b * KN + tid];
        float mx2 = pv;
        #pragma unroll
        for (int off = 16; off >= 1; off >>= 1) mx2 = fmaxf(mx2, __shfl_xor(mx2, off, 32));
        float e2 = __expf(pv - mx2);
        float s2 = e2;
        #pragma unroll
        for (int off = 16; off >= 1; off >>= 1) s2 += __shfl_xor(s2, off, 32);

        wmix[tid] = 0.5f * learned + 0.5f * (e2 / s2);
    }
    __syncthreads();

    // ---- pooling: out[b][d] = sum_k wmix[k] * nf32[k][d]  (d = tid)
    float acc = 0.0f;
    #pragma unroll 8
    for (int k = 0; k < KN; ++k) acc = fmaf(wmix[k], nf32[k][tid], acc);
    out[(long long)b * DIM + tid] = acc;
}

// ---------------------------------------------------------------------------
extern "C" void kernel_launch(void* const* d_in, const int* in_sizes, int n_in,
                              void* d_out, int out_size, void* d_ws, size_t ws_size,
                              hipStream_t stream) {
    const float*     x    = (const float*)d_in[0];
    const long long* nidx = (const long long*)d_in[1];   // int64 indices
    const float*     pwts = (const float*)d_in[2];
    const float*     W1   = (const float*)d_in[3];
    const float*     b1   = (const float*)d_in[4];
    const float*     W2   = (const float*)d_in[5];
    const float*     b2   = (const float*)d_in[6];
    float*           out  = (float*)d_out;

    _Float16* W1t = (_Float16*)d_ws;   // 128*256*2 = 64 KB scratch

    const int B = in_sizes[1] / KN;    // neighbor_idx flat count / K

    // Prepass: transpose + f32->f16 convert W1 (32768 elements)
    w1_to_f16_t_kernel<<<(DIM * HID) / 256, 256, 0, stream>>>(W1, W1t);

    // Main fused kernel: one block per batch element
    importance_pool_kernel<<<B, 256, 0, stream>>>(x, nidx, pwts, W1t,
                                                  b1, W2, b2, out);
}
